// EdgeClassifier_GAT_6820408066560
// MI455X (gfx1250) — compile-verified, matrix operand order checked by reading, not run on hardware
//
#include <hip/hip_runtime.h>
#include <math.h>

// ---------------------------------------------------------------------------
// EdgeClassifier GAT for MI455X (gfx1250, wave32, WMMA 16x16x32 f16)
//   N=50000 nodes, E=800000 edges, IN=128, HID=64, HEADS=4, OUT=2
//   All GEMM dims are multiples of 16/32 -> no bounds checks in WMMA kernels.
// ---------------------------------------------------------------------------

typedef _Float16 f16x16 __attribute__((ext_vector_type(16)));
typedef _Float16 f16x2  __attribute__((ext_vector_type(2)));
typedef float    f32x8  __attribute__((ext_vector_type(8)));

static constexpr int NN   = 50000;
static constexpr int EE   = 800000;
static constexpr int NEE  = EE + NN;   // edges + self loops
static constexpr int IN_C = 128;
static constexpr int HID  = 64;
static constexpr int HEADS= 4;
static constexpr int H1C  = HEADS * HID; // 256

// ---- CDNA5 16-bit A/B fragment K offset for VGPR i, lane-half h ------------
// A 16x32 f16 layout (ISA 7.12.2): lanes 0-15 rows M=0..15, VGPR0..3 hold
// K pairs 0..7, VGPR4..7 hold 16..23; lanes 16-31 same rows, K offset +8.
__device__ __forceinline__ int frag_koff(int i, int half) {
    return ((i & 3) << 1) + ((i >> 2) << 4) + (half << 3);
}

// Load a 16-element f16 fragment for row `row` of a row-major [*, ldk] matrix.
__device__ __forceinline__ f16x16 load_frag(const _Float16* base, int ldk,
                                            int row, int k0, int half) {
    f16x16 f;
    const _Float16* rp = base + (size_t)row * ldk + k0;
#pragma unroll
    for (int i = 0; i < 8; ++i) {
        f16x2 p = *(const f16x2*)(rp + frag_koff(i, half));
        f[2 * i]     = p[0];
        f[2 * i + 1] = p[1];
    }
    return f;
}

// ---------------------------------------------------------------------------
// Generic WMMA GEMM: D[M,Nn] = A[M,K] (f16) * Bt[Nn,K]^T (f16), f32 out.
// One wave per 16x16 output tile, K stepped by 32.
// ---------------------------------------------------------------------------
__global__ void wmma_gemm_kernel(const _Float16* __restrict__ A,
                                 const _Float16* __restrict__ Bt,
                                 float* __restrict__ D,
                                 int M, int Nn, int K) {
    int wave = (int)((blockIdx.x * blockDim.x + threadIdx.x) >> 5);
    int lane = threadIdx.x & 31;
    int ntiles = Nn >> 4;
    int total  = (M >> 4) * ntiles;
    if (wave >= total) return;                 // wave-uniform, EXEC stays full
    int mt = wave / ntiles, nt = wave - mt * ntiles;
    int m0 = mt << 4, n0 = nt << 4;
    int half = lane >> 4, r = lane & 15;

    f32x8 acc = {};
    for (int k0 = 0; k0 < K; k0 += 32) {
        f16x16 a = load_frag(A,  K, m0 + r, k0, half);
        f16x16 b = load_frag(Bt, K, n0 + r, k0, half);
        acc = __builtin_amdgcn_wmma_f32_16x16x32_f16(false, a, false, b,
                                                     (short)0, acc, false, false);
    }
    // C/D layout: lane half selects M base (0 / 8), VGPR i -> row i, col = r.
    int nn = n0 + r;
#pragma unroll
    for (int i = 0; i < 8; ++i)
        D[(size_t)(m0 + i + 8 * half) * Nn + nn] = acc[i];
}

// ---------------------------------------------------------------------------
// Elementwise helpers
// ---------------------------------------------------------------------------
__global__ void cast_f16_kernel(const float* __restrict__ in,
                                _Float16* __restrict__ out, int total) {
    int i = blockIdx.x * blockDim.x + threadIdx.x;
    if (i < total) out[i] = (_Float16)in[i];
}

// Wt[n*K + k] = (f16) W[k*Nn + n]
__global__ void transpose_cast_kernel(const float* __restrict__ W,
                                      _Float16* __restrict__ Wt, int K, int Nn) {
    int i = blockIdx.x * blockDim.x + threadIdx.x;
    if (i >= K * Nn) return;
    int k = i / Nn, n = i - k * Nn;
    Wt[(size_t)n * K + k] = (_Float16)W[i];
}

__global__ void fill_kernel(float* __restrict__ p, float v, int total) {
    int i = blockIdx.x * blockDim.x + threadIdx.x;
    if (i < total) p[i] = v;
}

// out[i] = act(in[i] + b[i % ld])  -> f16
__global__ void bias_act_cast_kernel(const float* __restrict__ in,
                                     const float* __restrict__ b,
                                     _Float16* __restrict__ out,
                                     int total, int ld, int do_relu) {
    int i = blockIdx.x * blockDim.x + threadIdx.x;
    if (i >= total) return;
    float v = in[i] + b[i % ld];
    if (do_relu) v = v > 0.f ? v : 0.f;
    out[i] = (_Float16)v;
}

// Per-node attention scores: ss/sd[n*Hh+h] = dot(H[n,h,:], a_{src,dst}[h,:])
__global__ void scores_kernel(const float* __restrict__ H,
                              const float* __restrict__ a_src,
                              const float* __restrict__ a_dst,
                              float* __restrict__ ss, float* __restrict__ sd,
                              int N, int Hh, int C) {
    int idx = blockIdx.x * blockDim.x + threadIdx.x;
    if (idx >= N * Hh) return;
    int n = idx / Hh, h = idx - n * Hh;
    const float* row = H + (size_t)n * Hh * C + (size_t)h * C;
    const float* as = a_src + h * C;
    const float* ad = a_dst + h * C;
    float s1 = 0.f, s2 = 0.f;
    for (int c = 0; c < C; ++c) { float v = row[c]; s1 += v * as[c]; s2 += v * ad[c]; }
    ss[idx] = s1;
    sd[idx] = s2;
}

// ---------------------------------------------------------------------------
// Edge softmax passes (segment max / sum / weighted scatter-add)
// ---------------------------------------------------------------------------
__device__ __forceinline__ float leaky02(float x) { return x > 0.f ? x : 0.2f * x; }

__device__ __forceinline__ void atomic_max_f32(float* addr, float val) {
    unsigned* ua = (unsigned*)addr;
    unsigned old = *ua;
    while (true) {
        float f = __uint_as_float(old);
        if (f >= val) break;
        unsigned assumed = old;
        old = atomicCAS(ua, assumed, __float_as_uint(val));
        if (old == assumed) break;
    }
}

__device__ __forceinline__ void edge_endpoints(const int* ei, int j,
                                               int& src, int& dst) {
    if (j < EE) { src = ei[j]; dst = ei[EE + j]; }
    else        { src = j - EE; dst = j - EE; }   // self loops
}

__global__ void edge_max_kernel(const int* __restrict__ ei,
                                const float* __restrict__ ss,
                                const float* __restrict__ sd,
                                float* __restrict__ m, int Hh) {
    int idx = blockIdx.x * blockDim.x + threadIdx.x;
    if (idx >= NEE * Hh) return;
    int j = idx / Hh, h = idx - j * Hh;
    int src, dst; edge_endpoints(ei, j, src, dst);
    float e = leaky02(ss[src * Hh + h] + sd[dst * Hh + h]);
    atomic_max_f32(&m[dst * Hh + h], e);
}

__global__ void edge_sum_kernel(const int* __restrict__ ei,
                                const float* __restrict__ ss,
                                const float* __restrict__ sd,
                                const float* __restrict__ m,
                                float* __restrict__ d, int Hh) {
    int idx = blockIdx.x * blockDim.x + threadIdx.x;
    if (idx >= NEE * Hh) return;
    int j = idx / Hh, h = idx - j * Hh;
    int src, dst; edge_endpoints(ei, j, src, dst);
    float e = leaky02(ss[src * Hh + h] + sd[dst * Hh + h]);
    atomicAdd(&d[dst * Hh + h], __expf(e - m[dst * Hh + h]));
}

// out[dst, h, :] += alpha * H[src, h, :]
__global__ void edge_agg_kernel(const int* __restrict__ ei,
                                const float* __restrict__ ss,
                                const float* __restrict__ sd,
                                const float* __restrict__ m,
                                const float* __restrict__ d,
                                const float* __restrict__ H,
                                float* __restrict__ out, int Hh, int C) {
    int idx = blockIdx.x * blockDim.x + threadIdx.x;
    if (idx >= NEE * Hh) return;
    int j = idx / Hh, h = idx - j * Hh;
    int src, dst; edge_endpoints(ei, j, src, dst);
    float e  = leaky02(ss[src * Hh + h] + sd[dst * Hh + h]);
    float ex = __expf(e - m[dst * Hh + h]);
    float alpha = ex / (d[dst * Hh + h] + 1e-16f);
    const float* hr = H   + (size_t)src * Hh * C + (size_t)h * C;
    float*       orow = out + (size_t)dst * Hh * C + (size_t)h * C;
    for (int c = 0; c < C; c += 4) {
        float4 v = *(const float4*)(hr + c);
        atomicAdd(orow + c + 0, v.x * alpha);
        atomicAdd(orow + c + 1, v.y * alpha);
        atomicAdd(orow + c + 2, v.z * alpha);
        atomicAdd(orow + c + 3, v.w * alpha);
    }
}

// ---------------------------------------------------------------------------
// Edge MLP: out[E,2] = relu(concat(h2[p], h2[c]) @ Wm1 + bm1) @ Wm2 + bm2
// Wave computes a 16-edge strip: gather-fused WMMA A load, 4 n-tiles of Wm1,
// relu+bias into LDS, then per-lane 64-length dot for the 64->2 layer.
// ---------------------------------------------------------------------------
__global__ void edge_mlp_kernel(const _Float16* __restrict__ h2h,
                                const int* __restrict__ ei,
                                const _Float16* __restrict__ Wm1t,  // [64,128]
                                const float* __restrict__ bm1,
                                const float* __restrict__ Wm2,      // [64,2]
                                const float* __restrict__ bm2,
                                float* __restrict__ out) {
    __shared__ float lds[8][16][HID + 1];
    int wid  = threadIdx.x >> 5;
    int lane = threadIdx.x & 31;
    int t    = blockIdx.x * 8 + wid;            // 16-edge tile, exact division
    int half = lane >> 4, r = lane & 15;

    int j = t * 16 + r;
    const _Float16* prow = h2h + (size_t)ei[j]      * HID;
    const _Float16* crow = h2h + (size_t)ei[EE + j] * HID;

    f32x8 acc[4] = {{}, {}, {}, {}};
    for (int ks = 0; ks < 4; ++ks) {
        int k0 = ks * 32;
        f16x16 a;
#pragma unroll
        for (int i = 0; i < 8; ++i) {
            int k = k0 + frag_koff(i, half);
            const _Float16* sp = (k < HID) ? (prow + k) : (crow + (k - HID));
            f16x2 p = *(const f16x2*)sp;
            a[2 * i]     = p[0];
            a[2 * i + 1] = p[1];
        }
#pragma unroll
        for (int nt = 0; nt < 4; ++nt) {
            f16x16 b = load_frag(Wm1t, 2 * HID, nt * 16 + r, k0, half);
            acc[nt] = __builtin_amdgcn_wmma_f32_16x16x32_f16(false, a, false, b,
                                                             (short)0, acc[nt],
                                                             false, false);
        }
    }
    // relu + bias into LDS
#pragma unroll
    for (int nt = 0; nt < 4; ++nt) {
        int col = nt * 16 + r;
        float bb = bm1[col];
#pragma unroll
        for (int i = 0; i < 8; ++i) {
            float g = acc[nt][i] + bb;
            lds[wid][i + 8 * half][col] = g > 0.f ? g : 0.f;
        }
    }
    __syncthreads();
    // 64 -> 2: lane handles (row = lane>>1, o = lane&1)
    int row = lane >> 1, o = lane & 1;
    float s = bm2[o];
#pragma unroll 8
    for (int n = 0; n < HID; ++n) s += lds[wid][row][n] * Wm2[n * 2 + o];
    out[(size_t)(t * 16 + row) * 2 + o] = s;
}

// ---------------------------------------------------------------------------
// Host orchestration
// ---------------------------------------------------------------------------
static inline int cdiv(long long a, int b) { return (int)((a + b - 1) / b); }

extern "C" void kernel_launch(void* const* d_in, const int* in_sizes, int n_in,
                              void* d_out, int out_size, void* d_ws, size_t ws_size,
                              hipStream_t stream) {
    (void)in_sizes; (void)n_in; (void)out_size; (void)ws_size;

    const float* x      = (const float*)d_in[0];
    const int*   ei     = (const int*)  d_in[1];   // [2,E] flat: [0..E)=src, [E..2E)=dst
    const float* W1     = (const float*)d_in[2];
    const float* a_src1 = (const float*)d_in[3];
    const float* a_dst1 = (const float*)d_in[4];
    const float* b1     = (const float*)d_in[5];
    const float* W2     = (const float*)d_in[6];
    const float* a_src2 = (const float*)d_in[7];
    const float* a_dst2 = (const float*)d_in[8];
    const float* b2     = (const float*)d_in[9];
    const float* Wm1    = (const float*)d_in[10];
    const float* bm1    = (const float*)d_in[11];
    const float* Wm2    = (const float*)d_in[12];
    const float* bm2    = (const float*)d_in[13];
    float* out = (float*)d_out;

    // bump allocator over workspace (~180 MB total)
    char* ws = (char*)d_ws;
    size_t off = 0;
    auto alloc = [&](size_t bytes) -> char* {
        char* p = ws + off;
        off = (off + bytes + 255) & ~(size_t)255;
        return p;
    };
    _Float16* xh    = (_Float16*)alloc((size_t)NN * IN_C * 2);
    _Float16* W1t   = (_Float16*)alloc((size_t)H1C * IN_C * 2);
    _Float16* W2t   = (_Float16*)alloc((size_t)HID * H1C * 2);
    _Float16* Wm1t  = (_Float16*)alloc((size_t)HID * 2 * HID * 2);
    float*    H1lin = (float*)   alloc((size_t)NN * H1C * 4);
    float*    ss1   = (float*)   alloc((size_t)NN * HEADS * 4);
    float*    sd1   = (float*)   alloc((size_t)NN * HEADS * 4);
    float*    m1    = (float*)   alloc((size_t)NN * HEADS * 4);
    float*    dn1   = (float*)   alloc((size_t)NN * HEADS * 4);
    float*    out1  = (float*)   alloc((size_t)NN * H1C * 4);
    _Float16* h1h   = (_Float16*)alloc((size_t)NN * H1C * 2);
    float*    H2lin = (float*)   alloc((size_t)NN * HID * 4);
    float*    ss2   = (float*)   alloc((size_t)NN * 4);
    float*    sd2   = (float*)   alloc((size_t)NN * 4);
    float*    m2    = (float*)   alloc((size_t)NN * 4);
    float*    dn2   = (float*)   alloc((size_t)NN * 4);
    float*    out2  = (float*)   alloc((size_t)NN * HID * 4);
    _Float16* h2h   = (_Float16*)alloc((size_t)NN * HID * 2);

    const int T = 256;

    // ---- prep: casts + weight transposes ----
    cast_f16_kernel<<<cdiv((long long)NN * IN_C, T), T, 0, stream>>>(x, xh, NN * IN_C);
    transpose_cast_kernel<<<cdiv(IN_C * H1C, T), T, 0, stream>>>(W1, W1t, IN_C, H1C);
    transpose_cast_kernel<<<cdiv(H1C * HID, T), T, 0, stream>>>(W2, W2t, H1C, HID);
    transpose_cast_kernel<<<cdiv(2 * HID * HID, T), T, 0, stream>>>(Wm1, Wm1t, 2 * HID, HID);

    // ---- layer 1: linear + attention ----
    wmma_gemm_kernel<<<cdiv((NN / 16) * (H1C / 16), 8), 256, 0, stream>>>(
        xh, W1t, H1lin, NN, H1C, IN_C);
    scores_kernel<<<cdiv(NN * HEADS, T), T, 0, stream>>>(
        H1lin, a_src1, a_dst1, ss1, sd1, NN, HEADS, HID);
    fill_kernel<<<cdiv(NN * HEADS, T), T, 0, stream>>>(m1, -3.0e38f, NN * HEADS);
    fill_kernel<<<cdiv(NN * HEADS, T), T, 0, stream>>>(dn1, 0.f, NN * HEADS);
    fill_kernel<<<cdiv((long long)NN * H1C, T), T, 0, stream>>>(out1, 0.f, NN * H1C);
    edge_max_kernel<<<cdiv((long long)NEE * HEADS, T), T, 0, stream>>>(ei, ss1, sd1, m1, HEADS);
    edge_sum_kernel<<<cdiv((long long)NEE * HEADS, T), T, 0, stream>>>(ei, ss1, sd1, m1, dn1, HEADS);
    edge_agg_kernel<<<cdiv((long long)NEE * HEADS, T), T, 0, stream>>>(
        ei, ss1, sd1, m1, dn1, H1lin, out1, HEADS, HID);
    bias_act_cast_kernel<<<cdiv((long long)NN * H1C, T), T, 0, stream>>>(
        out1, b1, h1h, NN * H1C, H1C, 1);

    // ---- layer 2: linear + attention (1 head, mean == identity) ----
    wmma_gemm_kernel<<<cdiv((NN / 16) * (HID / 16), 8), 256, 0, stream>>>(
        h1h, W2t, H2lin, NN, HID, H1C);
    scores_kernel<<<cdiv(NN, T), T, 0, stream>>>(
        H2lin, a_src2, a_dst2, ss2, sd2, NN, 1, HID);
    fill_kernel<<<cdiv(NN, T), T, 0, stream>>>(m2, -3.0e38f, NN);
    fill_kernel<<<cdiv(NN, T), T, 0, stream>>>(dn2, 0.f, NN);
    fill_kernel<<<cdiv((long long)NN * HID, T), T, 0, stream>>>(out2, 0.f, NN * HID);
    edge_max_kernel<<<cdiv(NEE, T), T, 0, stream>>>(ei, ss2, sd2, m2, 1);
    edge_sum_kernel<<<cdiv(NEE, T), T, 0, stream>>>(ei, ss2, sd2, m2, dn2, 1);
    edge_agg_kernel<<<cdiv(NEE, T), T, 0, stream>>>(
        ei, ss2, sd2, m2, dn2, H2lin, out2, 1, HID);
    bias_act_cast_kernel<<<cdiv((long long)NN * HID, T), T, 0, stream>>>(
        out2, b2, h2h, NN * HID, HID, 0);

    // ---- edge MLP (gather-fused WMMA + fused 64->2 epilogue) ----
    edge_mlp_kernel<<<(EE / 16) / 8, 256, 0, stream>>>(
        h2h, ei, Wm1t, bm1, Wm2, bm2, out);
}
